// kNNFit_71468255805744
// MI455X (gfx1250) — matrix operand order repaired and checked by reading, not compiled, not run on hardware
//
#include <hip/hip_runtime.h>
#include <hip/hip_bf16.h>

// ---------------------------------------------------------------------------
// Problem constants (fixed by the reference)
// ---------------------------------------------------------------------------
#define DD      512        // bit dimension
#define MPTS    131072     // stored points
#define BCOLS   64         // query batch
#define NT      5          // 4 query n-tiles + 1 tile whose col0 = W (gives s2)
#define NROWS   (NT * 16)  // 80 rows in the B operand matrix
#define KSTEP   32         // WMMA K per step (f16)
#define ROWS_PER_BLOCK 128 // 8 waves * 16 rows
#define NBLOCKS (MPTS / ROWS_PER_BLOCK)  // 1024
#define ROW_STRIDE 528     // padded halves per B row in LDS (512 + 16; TDM pad)

typedef __attribute__((ext_vector_type(16))) _Float16 v16h;
typedef __attribute__((ext_vector_type(8)))  _Float16 v8h;
typedef __attribute__((ext_vector_type(8)))  float    v8f;
typedef __attribute__((ext_vector_type(4)))  unsigned int u32x4;
typedef __attribute__((ext_vector_type(8)))  int          i32x8;
typedef __attribute__((ext_vector_type(4)))  int          i32x4;

#define LDS_B_BYTES   (NROWS * ROW_STRIDE * 2)      // 84480
#define LDS_SC_BYTES  (8 * BCOLS * 4)               // 2048
#define SMEM_BYTES    (LDS_B_BYTES + LDS_SC_BYTES)  // 86528  (WGP has 320KB)

// workspace layout (bytes)
#define WS_BMAT   0                       // 80*512 halves = 81920
#define WS_S1     81920                   // 64 floats
#define WS_NORMW  82176                   // 1 float
#define WS_PART   82432                   // 1024*64 floats = 262144

// ---------------------------------------------------------------------------
// two-stage rational squash F() from the reference
// ---------------------------------------------------------------------------
__device__ __forceinline__ float squashF(float x, float k, float j) {
    float inner = x * (1.0f - k) / (k * (1.0f - 2.0f * fabsf(x)) + 1.0f);
    float t = 2.0f * inner - 1.0f;
    return 0.5f + 0.5f * t * (1.0f + j) / (-j * (1.0f - 2.0f * fabsf(t)) + 1.0f);
}

// ---------------------------------------------------------------------------
// Kernel 1: build B operand matrix (f16), s1[b], ||W||_1 + eps
// ---------------------------------------------------------------------------
__global__ void knn_prep_kernel(const int* __restrict__ X1,
                                const float* __restrict__ W,
                                unsigned short* __restrict__ Bg_u16,
                                float* __restrict__ s1,
                                float* __restrict__ normW) {
    __shared__ float red[64];
    _Float16* Bg = reinterpret_cast<_Float16*>(Bg_u16);
    const int n = blockIdx.x;
    const int t = threadIdx.x;

    if (n < NROWS) {
        float part = 0.0f;
        for (int d = t; d < DD; d += 64) {
            float v;
            if (n < BCOLS)       v = W[d] * (float)X1[d * BCOLS + n];
            else if (n == BCOLS) v = W[d];
            else                 v = 0.0f;
            Bg[n * DD + d] = (_Float16)v;
            part += v;
        }
        if (n < BCOLS) {
            red[t] = part;
            __syncthreads();
            if (t == 0) {
                float s = 0.0f;
                for (int i = 0; i < 64; ++i) s += red[i];
                s1[n] = s;
            }
        }
    } else {
        float part = 0.0f;
        for (int d = t; d < DD; d += 64) part += fabsf(W[d]);
        red[t] = part;
        __syncthreads();
        if (t == 0) {
            float s = 0.0f;
            for (int i = 0; i < 64; ++i) s += red[i];
            normW[0] = s + 1e-6f;
        }
    }
}

// ---------------------------------------------------------------------------
// Kernel 2: main fused WMMA pass with TDM-staged B operand.
// ---------------------------------------------------------------------------
__global__ void __launch_bounds__(256, 2)
knn_main_kernel(const int* __restrict__ X2,
                const unsigned short* __restrict__ Bg_u16,
                const float* __restrict__ s1g,
                const float* __restrict__ normWp,
                const float* __restrict__ Y,
                const float* __restrict__ kp,
                const float* __restrict__ jp,
                float* __restrict__ partials) {
    extern __shared__ char smem[];
    _Float16* Bl = reinterpret_cast<_Float16*>(smem);
    float* scoreLDS = reinterpret_cast<float*>(smem + LDS_B_BYTES);

    const int tid    = threadIdx.x;
    const int wave   = tid >> 5;
    const int lane   = tid & 31;
    const int lane16 = lane & 15;
    const int hi     = lane >> 4;   // which half of the wave

    // ---- stage B operand matrix into padded LDS via Tensor Data Mover ----
    // 1D tensor: 80*512 f16 = 81920B contiguous in global; TDM pad inserts
    // 8 DWORDs (16 halves) after every 256 DWORDs (one 512-half row), which
    // reproduces the ROW_STRIDE=528 padded LDS layout in hardware.
    if (wave == 0) {
        const unsigned long long ga = (unsigned long long)Bg_u16;
        const unsigned lds_base = (unsigned)(size_t)(void*)smem; // flat[31:0] = LDS offset

        // D# group 0: count=1 | lds_addr | global_addr[56:0] | type=2
        u32x4 g0;
        g0.x = 1u;                                        // count=1, user mode
        g0.y = lds_base;                                  // LDS byte address
        g0.z = (unsigned)ga;                              // global_addr[31:0]
        g0.w = (unsigned)((ga >> 32) & 0x01FFFFFFu)       // global_addr[56:32]
             | 0x80000000u;                               // type=2 (bits 127:126)

        // D# group 1: data_size=2B, pad_enable, pad_interval=256DW, pad=8DW,
        // tensor_dim0 = tile_dim0 = 40960 elements, dim1 = 1.
        i32x8 g1;
        g1[0] = (int)((1u << 16)          // data_size = 1 -> 2 bytes
                    | (1u << 20)          // pad_enable
                    | (7u << 22)          // pad_interval code 7 = 256 DWORDs
                    | (7u << 25));        // pad_amount  code 7 = 8 DWORDs
        g1[1] = (int)(40960u << 16);      // tensor_dim0[15:0]=0xA000 (<<16)
        g1[2] = (int)(1u << 16);          // tensor_dim0[31:16]=0 | tensor_dim1=1
        g1[3] = (int)(40960u << 16);      // tile_dim0 = 0xA000
        g1[4] = 0;                        // tile_dim1=0 (unused), tile_dim2=0
        g1[5] = (int)40960u;              // tensor_dim0_stride[31:0]
        g1[6] = 0;
        g1[7] = 0;

        i32x4 z4 = {};
        i32x8 z8 = {};
        (void)z8;
#if defined(__clang_major__) && (__clang_major__ >= 23)
        __builtin_amdgcn_tensor_load_to_lds(g0, g1, z4, z4, z8, 0);
#else
        __builtin_amdgcn_tensor_load_to_lds(g0, g1, z4, z4, 0);
#endif
        __builtin_amdgcn_s_wait_tensorcnt(0);
    }
    __syncthreads();

    const int m0   = blockIdx.x * ROWS_PER_BLOCK + wave * 16;
    const int mrow = m0 + lane16;
    const int* xcol = X2 + mrow;        // column of X2 for this lane's row

    v8f acc[NT] = {};                   // f32 accumulators, 5 N-tiles

    // ---- K loop: D in steps of 32 ----
    #pragma unroll 2
    for (int d0 = 0; d0 < DD; d0 += KSTEP) {
        // A-matrix (16-bit, 16x32): lane holds row m=lane16; element e holds
        //   K(e) = e + 16*hi   (contiguous, matching B)
        // Bits are 0/1 -> pack two per DWORD: (b0 | b1<<16) * 0x3C00.
        union { v16h h; unsigned int u[8]; } A_;
        const int dbase = d0 + hi * 16;
        #pragma unroll
        for (int p = 0; p < 8; ++p) {
            const int e0  = 2 * p;
            const int dd0 = dbase + e0;
            const unsigned b0 = (unsigned)xcol[(size_t)dd0 * (size_t)MPTS];
            const unsigned b1 = (unsigned)xcol[(size_t)(dd0 + 1) * (size_t)MPTS];
            A_.u[p] = (b0 | (b1 << 16)) * 0x3C00u;
        }
        const v16h a = A_.h;

        if (d0 + KSTEP < DD) {
            __builtin_prefetch(xcol + (size_t)(d0 + KSTEP + hi * 8) * (size_t)MPTS, 0, 1);
        }

        // B-matrix (32x16 f16): lane = column N; lanes 0-15 K=0..15,
        // lanes 16-31 K=16..31 -> one contiguous 32B LDS load per N-tile.
        #pragma unroll
        for (int nt = 0; nt < NT; ++nt) {
            const _Float16* brow = &Bl[(nt * 16 + lane16) * ROW_STRIDE + d0 + hi * 16];
            v16h b = *reinterpret_cast<const v16h*>(brow);
            acc[nt] = __builtin_amdgcn_wmma_f32_16x16x32_f16(
                false, a, false, b, (short)0, acc[nt], false, false);
        }
    }

    // ---- epilogue ----
    // D-layout: lane l, vgpr r  ->  m_local = r + 8*hi, n = lane16.
    // s2[m] sits in column n==0 of tile 4: lane 0 (m=r) / lane 16 (m=8+r).
    const float kk = kp[0];
    const float jj = jp[0];
    const float invNorm = 1.0f / normWp[0];

    float s2v[8];
    #pragma unroll
    for (int r = 0; r < 8; ++r)
        s2v[r] = __shfl(acc[4][r], hi << 4, 32);   // ds_bpermute broadcast

    float yv[8];
    #pragma unroll
    for (int r = 0; r < 8; ++r)
        yv[r] = Y[m0 + hi * 8 + r];

    #pragma unroll
    for (int nt = 0; nt < 4; ++nt) {
        const float s1n = s1g[nt * 16 + lane16];
        float cs = 0.0f;
        #pragma unroll
        for (int r = 0; r < 8; ++r) {
            const float whd = s1n + s2v[r] - 2.0f * acc[nt][r];
            const float x   = 1.0f - whd * invNorm;
            cs += squashF(x, kk, jj) * yv[r];
        }
        cs += __shfl_xor(cs, 16, 32);              // fold the two m-halves
        if (lane < 16) scoreLDS[wave * BCOLS + nt * 16 + lane] = cs;
    }
    __syncthreads();

    if (tid < BCOLS) {
        float s = 0.0f;
        #pragma unroll
        for (int w = 0; w < 8; ++w) s += scoreLDS[w * BCOLS + tid];
        partials[blockIdx.x * BCOLS + tid] = s;    // deterministic partials
    }
}

// ---------------------------------------------------------------------------
// Kernel 3: fixed-order reduction over block partials -> beta*(mean + alpha)
// ---------------------------------------------------------------------------
__global__ void knn_reduce_kernel(const float* __restrict__ partials,
                                  const float* __restrict__ alpha,
                                  const float* __restrict__ beta,
                                  float* __restrict__ out) {
    const int b = threadIdx.x;   // 64 threads
    float s = 0.0f;
    for (int i = 0; i < NBLOCKS; ++i) s += partials[i * BCOLS + b];
    out[b] = beta[0] * (s * (1.0f / (float)MPTS) + alpha[0]);
}

// ---------------------------------------------------------------------------
// Host launcher
// ---------------------------------------------------------------------------
extern "C" void kernel_launch(void* const* d_in, const int* in_sizes, int n_in,
                              void* d_out, int out_size, void* d_ws, size_t ws_size,
                              hipStream_t stream) {
    const int*   X1    = (const int*)  d_in[0];   // [512,64]
    const int*   X2    = (const int*)  d_in[1];   // [512,131072]
    /* d_in[2] block-packed copy: unused */
    const float* Y     = (const float*)d_in[3];   // [131072]
    const float* W     = (const float*)d_in[4];   // [512]
    const float* alpha = (const float*)d_in[5];
    const float* beta  = (const float*)d_in[6];
    const float* kp    = (const float*)d_in[7];
    const float* jp    = (const float*)d_in[8];
    float* out = (float*)d_out;

    char* ws = (char*)d_ws;
    unsigned short* Bg      = (unsigned short*)(ws + WS_BMAT);
    float*          s1      = (float*)(ws + WS_S1);
    float*          normW   = (float*)(ws + WS_NORMW);
    float*          partial = (float*)(ws + WS_PART);

    knn_prep_kernel<<<NROWS + 1, 64, 0, stream>>>(X1, W, Bg, s1, normW);
    knn_main_kernel<<<NBLOCKS, 256, SMEM_BYTES, stream>>>(
        X2, Bg, s1, normW, Y, kp, jp, partial);
    knn_reduce_kernel<<<1, BCOLS, 0, stream>>>(partial, alpha, beta, out);
}
